// DiTMoELayer_8761733284136
// MI455X (gfx1250) — matrix operand order repaired and checked by reading, not compiled, not run on hardware
//
#include <hip/hip_runtime.h>

#define BB 2
#define SS 2048
#define HH 1024
#define DD 2048
#define EE 8
#define TOKENS (BB * SS)   // 4096
#define KC 32              // WMMA K depth per chunk

typedef __attribute__((ext_vector_type(16))) __bf16 v16bf;
typedef __attribute__((ext_vector_type(8)))  float  v8f;

union Frag { v16bf v; uint4 q[2]; };

__device__ __forceinline__ unsigned short f2bf(float f) {
    unsigned int u = __float_as_uint(f);
    unsigned int r = (u + 0x7FFFu + ((u >> 16) & 1u)) >> 16;  // RNE
    return (unsigned short)r;
}
__device__ __forceinline__ unsigned int pk2(float a, float b) {
    return (unsigned int)f2bf(a) | ((unsigned int)f2bf(b) << 16);
}

// ---------------- zero out + counters ----------------
__global__ void k_zero(float* __restrict__ out, int n4,
                       int* __restrict__ cnt, float* __restrict__ sump) {
    int i = blockIdx.x * blockDim.x + threadIdx.x;
    if (i < n4) ((float4*)out)[i] = make_float4(0.f, 0.f, 0.f, 0.f);
    if (i < EE) { cnt[i] = 0; sump[i] = 0.f; }
}

// ---------------- x -> bf16 ----------------
__global__ void k_cvt_x(const float* __restrict__ x, unsigned short* __restrict__ xb, int n8) {
    int i = blockIdx.x * blockDim.x + threadIdx.x;
    if (i >= n8) return;
    const float4* s = (const float4*)(x + (size_t)i * 8);
    float4 a = s[0], b = s[1];
    *(uint4*)(xb + (size_t)i * 8) =
        make_uint4(pk2(a.x, a.y), pk2(a.z, a.w), pk2(b.x, b.y), pk2(b.z, b.w));
}

// -------- W[e][K][N] f32 -> WT chunked-transposed bf16: WT[((e*KD/32+kb)*ND+n)*32+k] ----
__global__ __launch_bounds__(256) void k_cvt_w(const float* __restrict__ W,
                                               unsigned short* __restrict__ WT,
                                               int KD, int ND) {
    __shared__ float tile[32][33];
    const int e = blockIdx.z;
    const int k0 = blockIdx.x * 32;
    const int n0 = blockIdx.y * 32;
    const float* We = W + (size_t)e * KD * ND;
    const int t = threadIdx.x;
    const int r = t >> 3, c4 = (t & 7) * 4;
    float4 v = *(const float4*)(We + (size_t)(k0 + r) * ND + n0 + c4);
    tile[r][c4] = v.x; tile[r][c4 + 1] = v.y; tile[r][c4 + 2] = v.z; tile[r][c4 + 3] = v.w;
    __syncthreads();
    unsigned short* dst = WT + ((size_t)(e * (KD / 32) + blockIdx.x) * ND + n0 + r) * 32 + c4;
    *(uint2*)dst = make_uint2(pk2(tile[c4][r], tile[c4 + 1][r]),
                              pk2(tile[c4 + 2][r], tile[c4 + 3][r]));
}

// ---------------- router: logits, softmax, top-2, lists, aux partials ----
__global__ __launch_bounds__(256) void k_router(
    const float* __restrict__ x, const float* __restrict__ gw,
    int* __restrict__ cnt, float* __restrict__ sump,
    int* __restrict__ toks, float* __restrict__ wts) {
    __shared__ float sp[EE];
    const int tid = threadIdx.x;
    const int lane = tid & 31, wave = tid >> 5;
    const int t = blockIdx.x * 8 + wave;
    if (tid < EE) sp[tid] = 0.f;
    __syncthreads();

    float acc[EE];
#pragma unroll
    for (int e = 0; e < EE; ++e) acc[e] = 0.f;
    const float* xr = x + (size_t)t * HH;
    for (int j = lane; j < HH; j += 32) {
        float xv = xr[j];
        const float* g = gw + (size_t)j * EE;
#pragma unroll
        for (int e = 0; e < EE; ++e) acc[e] += xv * g[e];
    }
#pragma unroll
    for (int e = 0; e < EE; ++e) {
#pragma unroll
        for (int off = 16; off > 0; off >>= 1)
            acc[e] += __shfl_xor(acc[e], off, 32);
    }
    if (lane == 0) {
        float m = acc[0];
#pragma unroll
        for (int e = 1; e < EE; ++e) m = fmaxf(m, acc[e]);
        float ex[EE]; float s = 0.f;
#pragma unroll
        for (int e = 0; e < EE; ++e) { ex[e] = expf(acc[e] - m); s += ex[e]; }
        const float inv = 1.f / s;
        int e1 = 0;
#pragma unroll
        for (int e = 1; e < EE; ++e) if (ex[e] > ex[e1]) e1 = e;
        int e2 = (e1 == 0) ? 1 : 0;
#pragma unroll
        for (int e = 0; e < EE; ++e) if (e != e1 && ex[e] > ex[e2]) e2 = e;
        const float p1 = ex[e1], p2 = ex[e2], d = p1 + p2;
        int pos1 = atomicAdd(&cnt[e1], 1);
        toks[e1 * TOKENS + pos1] = t; wts[e1 * TOKENS + pos1] = p1 / d;
        int pos2 = atomicAdd(&cnt[e2], 1);
        toks[e2 * TOKENS + pos2] = t; wts[e2 * TOKENS + pos2] = p2 / d;
#pragma unroll
        for (int e = 0; e < EE; ++e) atomicAdd(&sp[e], ex[e] * inv);
    }
    __syncthreads();
    if (tid < EE) atomicAdd(&sump[tid], sp[tid]);
}

// ---------------- prefix offsets + aux loss ----------------
__global__ void k_finalize(const int* __restrict__ cnt, const float* __restrict__ sump,
                           int* __restrict__ offs, float* __restrict__ aux_out) {
    int off = 0; float s = 0.f;
    for (int e = 0; e < EE; ++e) {
        offs[e] = off; off += cnt[e];
        s += ((float)cnt[e] / (float)BB) * (sump[e] / (float)(BB * SS));
    }
    aux_out[0] = (float)EE * s;
}

// ---------------- expert GEMM (both passes), bf16 WMMA ----------------
// FIRST: h = gelu(gather(xb) @ W1[e] + b1)  -> hbuf (bf16, grouped by expert)
// !FIRST: out += w * (h @ W2[e] + b2)       -> atomic scatter
template <bool FIRST, int KDIM, int NDIM>
__global__ __launch_bounds__(256) void k_ffn(
    const unsigned short* __restrict__ xb, const unsigned short* __restrict__ WT,
    const float* __restrict__ bias, const int* __restrict__ toks,
    const float* __restrict__ wts, const int* __restrict__ cnt,
    const int* __restrict__ offs, unsigned short* __restrict__ hbuf,
    float* __restrict__ out) {
    const int e  = blockIdx.z;
    const int m0 = blockIdx.y * 128;
    const int n0 = blockIdx.x * 128;
    const int count = cnt[e];
    if (m0 >= count) return;
    const int hoff = offs[e];

    __shared__ __align__(16) unsigned short As[2][128][40];  // [row][K] + pad
    __shared__ __align__(16) unsigned short Bs[2][128][40];  // [N][K] + pad
    __shared__ int   toks_s[128];
    __shared__ float wts_s[128];

    const int tid = threadIdx.x;
    const int lane = tid & 31, wv = tid >> 5;
    const int wm = wv >> 2, wn = wv & 3;        // 2x4 wave grid, wave tile 64x32
    const int lidx = lane & 15, lhalf = lane >> 4;
    const int ar = tid >> 1, ah = tid & 1;      // A staging: row, 16-elem half
    const int bn2 = tid >> 1, bh = tid & 1;     // B staging: N row, 16-elem half

    if (tid < 128) {
        int r = m0 + tid;
        toks_s[tid] = (r < count) ? toks[e * TOKENS + r] : 0;
        wts_s[tid]  = (r < count) ? wts[e * TOKENS + r] : 0.f;
    }
    __syncthreads();

    uint4 rA0, rA1, rB0, rB1;  // packed bf16 staging regs

    auto loadA = [&](int kb) {
        if constexpr (FIRST) {
            const uint4* src = (const uint4*)(xb + (size_t)toks_s[ar] * HH +
                                              (size_t)kb * KC + ah * 16);
            rA0 = src[0]; rA1 = src[1];
        } else {
            if (m0 + ar < count) {
                const uint4* src = (const uint4*)(hbuf + (size_t)(hoff + m0 + ar) * DD +
                                                  (size_t)kb * KC + ah * 16);
                rA0 = src[0]; rA1 = src[1];
            } else {
                rA0 = make_uint4(0u, 0u, 0u, 0u); rA1 = rA0;
            }
        }
    };
    auto loadB = [&](int kb) {  // chunked-transposed weights: fully coalesced b128s
        const uint4* src = (const uint4*)(WT +
            ((size_t)(e * (KDIM / 32) + kb) * NDIM + n0 + bn2) * 32 + bh * 16);
        rB0 = src[0]; rB1 = src[1];
    };
    auto storeA = [&](int buf) {
        uint4* dst = (uint4*)&As[buf][ar][ah * 16];
        dst[0] = rA0; dst[1] = rA1;
    };
    auto storeB = [&](int buf) {
        uint4* dst = (uint4*)&Bs[buf][bn2][bh * 16];
        dst[0] = rB0; dst[1] = rB1;
    };

    v8f c[4][2];
#pragma unroll
    for (int i = 0; i < 4; ++i)
#pragma unroll
        for (int j = 0; j < 2; ++j)
#pragma unroll
            for (int r = 0; r < 8; ++r) c[i][j][r] = 0.f;

    auto compute = [&](int buf) {
        Frag b[2];
#pragma unroll
        for (int j = 0; j < 2; ++j) {
            const int n = wn * 32 + j * 16 + lidx;
            b[j].q[0] = *(const uint4*)&Bs[buf][n][lhalf * 8];
            b[j].q[1] = *(const uint4*)&Bs[buf][n][16 + lhalf * 8];
        }
#pragma unroll
        for (int i = 0; i < 4; ++i) {
            Frag a;
            const int m = wm * 64 + i * 16 + lidx;
            a.q[0] = *(const uint4*)&As[buf][m][lhalf * 8];
            a.q[1] = *(const uint4*)&As[buf][m][16 + lhalf * 8];
#pragma unroll
            for (int j = 0; j < 2; ++j)
                c[i][j] = __builtin_amdgcn_wmma_f32_16x16x32_bf16(
                    false, a.v, false, b[j].v, (short)0, c[i][j], false, false);
        }
    };

    const int KSTEPS = KDIM / KC;
    loadA(0); loadB(0);
    storeA(0); storeB(0);
    __syncthreads();
#pragma unroll 1
    for (int kb = 0; kb < KSTEPS; ++kb) {
        const int cur = kb & 1;
        if (kb + 1 < KSTEPS) { loadA(kb + 1); loadB(kb + 1); }
        compute(cur);
        if (kb + 1 < KSTEPS) { storeA(cur ^ 1); storeB(cur ^ 1); }
        __syncthreads();
    }

    const float* be = bias + (size_t)e * NDIM;
#pragma unroll
    for (int i = 0; i < 4; ++i)
#pragma unroll
        for (int j = 0; j < 2; ++j)
#pragma unroll
            for (int r = 0; r < 8; ++r) {
                const int ml = wm * 64 + i * 16 + lhalf * 8 + r;
                const int gm = m0 + ml;
                if (gm < count) {
                    const int gn = n0 + wn * 32 + j * 16 + lidx;
                    float v = c[i][j][r] + be[gn];
                    if constexpr (FIRST) {
                        float g = 0.5f * v * (1.f + erff(v * 0.70710678118654752f));
                        hbuf[(size_t)(hoff + gm) * DD + gn] = f2bf(g);
                    } else {
                        atomicAdd(&out[(size_t)toks_s[ml] * HH + gn], wts_s[ml] * v);
                    }
                }
            }
}

extern "C" void kernel_launch(void* const* d_in, const int* in_sizes, int n_in,
                              void* d_out, int out_size, void* d_ws, size_t ws_size,
                              hipStream_t stream) {
    const float* x  = (const float*)d_in[0];
    const float* gw = (const float*)d_in[1];
    const float* W1 = (const float*)d_in[2];
    const float* b1 = (const float*)d_in[3];
    const float* W2 = (const float*)d_in[4];
    const float* b2 = (const float*)d_in[5];
    float* out = (float*)d_out;

    char* ws = (char*)d_ws;
    int*   cnt  = (int*)(ws + 0);
    float* sump = (float*)(ws + 32);
    int*   offs = (int*)(ws + 64);
    int*   toks = (int*)(ws + 128);
    float* wts  = (float*)(ws + 128 + 4 * EE * TOKENS);
    size_t off = 128 + 8 * (size_t)EE * TOKENS;              // 262272
    unsigned short* xb   = (unsigned short*)(ws + off); off += (size_t)TOKENS * HH * 2;
    unsigned short* w1t  = (unsigned short*)(ws + off); off += (size_t)EE * HH * DD * 2;
    unsigned short* w2t  = (unsigned short*)(ws + off); off += (size_t)EE * DD * HH * 2;
    unsigned short* hbuf = (unsigned short*)(ws + off);      // 2*TOKENS x DD bf16

    const int n4 = TOKENS * HH / 4;
    k_zero<<<(n4 + 255) / 256, 256, 0, stream>>>(out, n4, cnt, sump);
    k_cvt_x<<<(TOKENS * HH / 8 + 255) / 256, 256, 0, stream>>>(x, xb, TOKENS * HH / 8);
    dim3 gw1(HH / 32, DD / 32, EE);
    k_cvt_w<<<gw1, 256, 0, stream>>>(W1, w1t, HH, DD);
    dim3 gw2(DD / 32, HH / 32, EE);
    k_cvt_w<<<gw2, 256, 0, stream>>>(W2, w2t, DD, HH);
    k_router<<<TOKENS / 8, 256, 0, stream>>>(x, gw, cnt, sump, toks, wts);
    k_finalize<<<1, 1, 0, stream>>>(cnt, sump, offs, out + (size_t)TOKENS * HH);

    dim3 g1(DD / 128, TOKENS / 128, EE);
    k_ffn<true, HH, DD><<<g1, 256, 0, stream>>>(xb, w1t, b1, toks, wts, cnt, offs, hbuf, out);
    dim3 g2(HH / 128, TOKENS / 128, EE);
    k_ffn<false, DD, HH><<<g2, 256, 0, stream>>>(xb, w2t, b2, toks, wts, cnt, offs, hbuf, out);
}
